// DGCNN_sample_52913997087296
// MI455X (gfx1250) — matrix operand order repaired and checked by reading, not compiled, not run on hardware
//
#include <hip/hip_runtime.h>

typedef __attribute__((ext_vector_type(16))) _Float16 v16h;
typedef __attribute__((ext_vector_type(8)))  float    v8f;

#define EPSBN 1e-5f
#define KNBR  50

// ---------------- WMMA fragment helpers (CDNA5 16x16x32 f16 layouts) ----------------
// A: 16x32 f16. lane<16 holds M=lane, K=0..15; lane>=16 holds M=lane-16, K=16..31
__device__ inline v16h frag_a(const _Float16* w, int ld, int row0, int kbase, int lane) {
  int r = lane & 15, hi = (lane >> 4) & 1;
  const _Float16* p = w + (size_t)(row0 + r) * ld + kbase + hi * 16;
  v16h a;
#pragma unroll
  for (int j = 0; j < 8; ++j) a[j] = p[j];
#pragma unroll
  for (int j = 0; j < 8; ++j) a[8 + j] = p[8 + j];
  return a;
}

// Fragment-ready B storage: for K-step ks, column c (0..127), half hi, the 16 halves a lane
// reads are contiguous. Index of element (k, c) with k global:
__device__ inline int fb_index(int k, int c) {
  return (((((k >> 5) << 7) + c) << 1) + ((k >> 4) & 1)) * 16 + (k & 15);
}
// B: 32x16 f16. lanes 0-15: N=col0+lane, K=ks*32+0..15 ; lanes 16-31: N=col0+lane-16, K=ks*32+16..31
__device__ inline v16h frag_b_fr(const _Float16* f, int ks, int col0, int lane) {
  const _Float16* p = f + ((((ks << 7) + col0 + (lane & 15)) << 1) + ((lane >> 4) & 1)) * 16;
  v16h b;
#pragma unroll
  for (int j = 0; j < 16; ++j) b[j] = p[j];
  return b;
}
__device__ inline v8f wmma16(v16h a, v16h b, v8f c) {
  return __builtin_amdgcn_wmma_f32_16x16x32_f16(false, a, false, b, (short)0, c, false, false);
}
__device__ inline float lrelu(float x) { return x >= 0.f ? x : 0.2f * x; }

// ---------------- kNN on grid (M x M, top-50 smallest sqdist) ----------------
__global__ void knn_kernel(const float* __restrict__ gf, int C, int Mm, int total,
                           int* __restrict__ nbr) {
  int gid = blockIdx.x * blockDim.x + threadIdx.x;
  if (gid >= total) return;
  int b = gid / Mm, i = gid % Mm;
  const float* base = gf + (size_t)b * C * Mm;
  float vloc[64];
  for (int c = 0; c < C; ++c) vloc[c] = base[(size_t)c * Mm + i];
  float ni = 0.f;
  for (int c = 0; c < C; ++c) ni += vloc[c] * vloc[c];
  float bestd[KNBR]; int besti[KNBR];
  for (int k = 0; k < KNBR; ++k) { bestd[k] = 3.4e38f; besti[k] = 0; }
  for (int j = 0; j < Mm; ++j) {
    float nj = 0.f, dot = 0.f;
    for (int c = 0; c < C; ++c) {
      float vj = base[(size_t)c * Mm + j];
      nj += vj * vj; dot += vloc[c] * vj;
    }
    float d = ni + nj - 2.f * dot;
    if (d < bestd[KNBR - 1]) {
      int pos = KNBR - 1;
      while (pos > 0 && bestd[pos - 1] > d) {
        bestd[pos] = bestd[pos - 1]; besti[pos] = besti[pos - 1]; --pos;
      }
      bestd[pos] = d; besti[pos] = j;
    }
  }
  for (int k = 0; k < KNBR; ++k) nbr[(size_t)gid * KNBR + k] = besti[k];
}

// ---------------- nearest grid point per point ----------------
__global__ void nearest_kernel(const float* __restrict__ ptf, const float* __restrict__ gf,
                               int C, int Nn, int Mm, int total, int* __restrict__ nearest) {
  int gid = blockIdx.x * blockDim.x + threadIdx.x;
  if (gid >= total) return;
  int b = gid / Nn, p = gid % Nn;
  float vloc[64];
  for (int c = 0; c < C; ++c) vloc[c] = ptf[((size_t)b * C + c) * Nn + p];
  float best = 3.4e38f; int bi = 0;
  for (int j = 0; j < Mm; ++j) {
    float d = 0.f;
    for (int c = 0; c < C; ++c) {
      float diff = vloc[c] - gf[((size_t)b * C + c) * Mm + j];
      d += diff * diff;
    }
    if (d < best) { best = d; bi = j; }
  }
  nearest[gid] = bi;
}

// ---------------- fused edge-conv (gather + up to 2 WMMA GEMMs + stats/max) ----------------
// Block = 256 threads (8 waves), covers 2 points x 64 padded-k columns = 128 cols.
template <int CIN, int KD, bool SECOND, bool FINAL>
__global__ void __launch_bounds__(256)
edgeconv_kernel(const float* __restrict__ ptf, const float* __restrict__ gridf,
                const int* __restrict__ nearest, const int* __restrict__ nbr,
                const float* __restrict__ WA, const float* __restrict__ WB,
                const float* __restrict__ sbA, const float* __restrict__ sbB,
                float* __restrict__ partS, float* __restrict__ partQ,
                float* __restrict__ xout, int Nn, int Mm) {
  // 32KB region: holds f16 frag-ready feature tiles (KD x 128, then F2 64 x 128)
  // and is later reused as the 64 x 128 f32 output tile for the epilogue.
  __shared__ __attribute__((aligned(16))) float FbufF[64 * 128];
  _Float16* Fbuf = (_Float16*)FbufF;
  float* Yt = FbufF;
  __shared__ __attribute__((aligned(16))) _Float16 Wl[64 * KD];
  __shared__ __attribute__((aligned(16))) _Float16 W2l[64 * 64];
  __shared__ float red[256];
  __shared__ float red2[256];
  __shared__ int gidx[128];

  const int tid = threadIdx.x;
  const int lane = tid & 31, wave = tid >> 5;
  const int npairs = Nn >> 1;
  const int b = blockIdx.x / npairs;
  const int n0 = (blockIdx.x % npairs) * 2;

  for (int i = tid; i < 64 * KD; i += 256) {
    int k = i % KD;
    Wl[i] = (k < 2 * CIN) ? (_Float16)WA[(i / KD) * (2 * CIN) + k] : (_Float16)0.f;
  }
  if (SECOND)
    for (int i = tid; i < 64 * 64; i += 256) W2l[i] = (_Float16)WB[i];

  if (tid < 128) {
    int p = n0 + (tid >> 6), kk = tid & 63;
    if (kk >= KNBR) kk = 0;   // pad columns duplicate the k=0 column
    gidx[tid] = (kk == 0) ? -1
        : nbr[((size_t)b * Mm + nearest[(size_t)b * Nn + p]) * KNBR + kk];
  }
  __syncthreads();

  // Build feature tile (frag-ready): rows [0,CIN)=nbr-ctr, [CIN,2CIN)=ctr, rest zero.
  for (int i = tid; i < KD * 128; i += 256) {
    int r = i >> 7, c = i & 127;
    float v = 0.f;
    if (r < 2 * CIN) {
      int p = n0 + (c >> 6);
      int cc = (r < CIN) ? r : r - CIN;
      float ctr = ptf[((size_t)b * CIN + cc) * Nn + p];
      if (r >= CIN) v = ctr;
      else {
        int g = gidx[c];
        v = (g < 0) ? 0.f : (gridf[((size_t)b * CIN + r) * Mm + g] - ctr);
      }
    }
    Fbuf[fb_index(r, c)] = (_Float16)v;
  }
  __syncthreads();

  const int cb = wave * 16;
  v8f acc[4];
#pragma unroll
  for (int mt = 0; mt < 4; ++mt)
#pragma unroll
    for (int j = 0; j < 8; ++j) acc[mt][j] = 0.f;

#pragma unroll
  for (int ks = 0; ks < KD / 32; ++ks) {
    v16h bf = frag_b_fr(Fbuf, ks, cb, lane);
#pragma unroll
    for (int mt = 0; mt < 4; ++mt) {
      v16h af = frag_a(Wl, KD, mt * 16, ks * 32, lane);
      acc[mt] = wmma16(af, bf, acc[mt]);
    }
  }
  __syncthreads();   // feature tile dead now

  const int hi = (lane >> 4) & 1;
  const int ln = lane & 15;
  const int col = cb + ln;

  if (SECOND) {
    // BN(convA) + lrelu -> F2 (f16, frag-ready) in Fbuf, then GEMM with W2
#pragma unroll
    for (int mt = 0; mt < 4; ++mt)
#pragma unroll
      for (int r = 0; r < 8; ++r) {
        int row = mt * 16 + r + hi * 8;
        float v = lrelu(acc[mt][r] * sbA[row] + sbA[64 + row]);
        Fbuf[fb_index(row, col)] = (_Float16)v;
      }
    __syncthreads();
    v8f acc2[4];
#pragma unroll
    for (int mt = 0; mt < 4; ++mt)
#pragma unroll
      for (int j = 0; j < 8; ++j) acc2[mt][j] = 0.f;
#pragma unroll
    for (int ks = 0; ks < 2; ++ks) {
      v16h bf = frag_b_fr(Fbuf, ks, cb, lane);
#pragma unroll
      for (int mt = 0; mt < 4; ++mt) {
        v16h af = frag_a(W2l, 64, mt * 16, ks * 32, lane);
        acc2[mt] = wmma16(af, bf, acc2[mt]);
      }
    }
#pragma unroll
    for (int mt = 0; mt < 4; ++mt) acc[mt] = acc2[mt];
    __syncthreads();   // F2 reads done; Yt may overwrite region
  }

  const float* sbL = SECOND ? sbB : sbA;
  if (FINAL) {
    // BN(last) + lrelu into f32 LDS tile, then max over real k columns -> x_out
#pragma unroll
    for (int mt = 0; mt < 4; ++mt)
#pragma unroll
      for (int r = 0; r < 8; ++r) {
        int row = mt * 16 + r + hi * 8;
        Yt[row * 128 + col] = lrelu(acc[mt][r] * sbL[row] + sbL[64 + row]);
      }
    __syncthreads();
    if (tid < 128) {
      int row = tid & 63, pt = tid >> 6;
      float m = -3.4e38f;
      for (int c = pt * 64; c < pt * 64 + KNBR; ++c) {
        float v = Yt[row * 128 + c];
        m = m > v ? m : v;
      }
      xout[((size_t)b * 64 + row) * Nn + n0 + pt] = m;
    }
  } else {
    // raw output tile -> deterministic per-block stat partials
#pragma unroll
    for (int mt = 0; mt < 4; ++mt)
#pragma unroll
      for (int r = 0; r < 8; ++r) {
        int row = mt * 16 + r + hi * 8;
        Yt[row * 128 + col] = acc[mt][r];
      }
    __syncthreads();
    {
      int row = tid >> 2, sub = tid & 3;
      float s = 0.f, q = 0.f;
      int c0 = sub * 32;
      for (int c = c0; c < c0 + 32; ++c) {
        if ((c & 63) < KNBR) {
          float v = Yt[row * 128 + c];
          s += v; q += v * v;
        }
      }
      red[tid] = s; red2[tid] = q;
    }
    __syncthreads();
    if (tid < 64) {
      float s = red[tid * 4] + red[tid * 4 + 1] + red[tid * 4 + 2] + red[tid * 4 + 3];
      float q = red2[tid * 4] + red2[tid * 4 + 1] + red2[tid * 4 + 2] + red2[tid * 4 + 3];
      partS[(size_t)tid * gridDim.x + blockIdx.x] = s;
      partQ[(size_t)tid * gridDim.x + blockIdx.x] = q;
    }
  }
}

// ---------------- stat reduce -> per-channel scale/bias ----------------
__global__ void reduce_stats_kernel(const float* __restrict__ pS, const float* __restrict__ pQ,
                                    int nblk, float cnt, const float* __restrict__ gamma,
                                    const float* __restrict__ beta, float* __restrict__ sb,
                                    int nch) {
  int ch = blockIdx.x * blockDim.x + threadIdx.x;
  if (ch >= nch) return;
  float s = 0.f, q = 0.f;
  for (int i = 0; i < nblk; ++i) { s += pS[(size_t)ch * nblk + i]; q += pQ[(size_t)ch * nblk + i]; }
  float mean = s / cnt;
  float var = q / cnt - mean * mean;
  float inv = gamma[ch] * rsqrtf(var + EPSBN);
  sb[ch] = inv;
  sb[nch + ch] = beta[ch] - mean * inv;
}

// ---------------- FPS gather: xg[b][c][m] = x[b][c][FPS[b][m]] ----------------
__global__ void fps_gather_kernel(const float* __restrict__ x, const int* __restrict__ fps,
                                  float* __restrict__ xg, int C, int Nn, int Mm, int total) {
  int gid = blockIdx.x * blockDim.x + threadIdx.x;
  if (gid >= total) return;
  int m = gid % Mm;
  int t = gid / Mm;
  int b = t / C;
  xg[gid] = x[(size_t)t * Nn + fps[(size_t)b * Mm + m]];
}

// ---------------- concat builders (to f16 GEMM inputs) ----------------
__global__ void build_xg_kernel(const float* __restrict__ x1g, const float* __restrict__ x2g,
                                const float* __restrict__ x3g, _Float16* __restrict__ xg,
                                int Mm, int Bb) {
  int cols = Bb * Mm;
  int total = 192 * cols;
  int gid = blockIdx.x * blockDim.x + threadIdx.x;
  if (gid >= total) return;
  int row = gid / cols, col = gid % cols;
  int b = col / Mm, m = col % Mm;
  const float* src = (row < 64) ? x1g : (row < 128 ? x2g : x3g);
  int c = row & 63;
  xg[gid] = (_Float16)src[((size_t)b * 64 + c) * Mm + m];
}

__global__ void max_m_kernel(const float* __restrict__ y6, float* __restrict__ xgmax,
                             int Mm, int Bb, int O) {
  int gid = blockIdx.x * blockDim.x + threadIdx.x;
  if (gid >= Bb * O) return;
  int b = gid / O, ch = gid % O;
  float m = -3.4e38f;
  for (int j = 0; j < Mm; ++j) {
    float v = y6[(size_t)ch * (Bb * Mm) + b * Mm + j];
    m = m > v ? m : v;
  }
  xgmax[gid] = m;
}

__global__ void build_hin_kernel(const float* __restrict__ xgmax, const float* __restrict__ x1,
                                 const float* __restrict__ x2, const float* __restrict__ x3,
                                 _Float16* __restrict__ hin, int Nn, int Bb) {
  int cols = Bb * Nn;
  int total = 704 * cols;
  int gid = blockIdx.x * blockDim.x + threadIdx.x;
  if (gid >= total) return;
  int row = gid / cols, col = gid % cols;
  int b = col / Nn, n = col % Nn;
  float v;
  if (row < 512)      v = xgmax[(size_t)b * 512 + row];
  else if (row < 576) v = x1[((size_t)b * 64 + row - 512) * Nn + n];
  else if (row < 640) v = x2[((size_t)b * 64 + row - 576) * Nn + n];
  else                v = x3[((size_t)b * 64 + row - 640) * Nn + n];
  hin[gid] = (_Float16)v;
}

// ---------------- generic WMMA GEMM for conv1 layers ----------------
// MODE 0: stats partials; 1: BN+lrelu -> f16 out; 2: BN+lrelu -> f32 out; 3: plain -> transposed f32 out
template <int MODE>
__global__ void __launch_bounds__(256)
conv1_kernel(const float* __restrict__ W, const _Float16* __restrict__ X,
             int C, int cols, int O, const float* __restrict__ sb,
             float* __restrict__ partS, float* __restrict__ partQ,
             _Float16* __restrict__ outH, float* __restrict__ outF) {
  __shared__ __attribute__((aligned(16))) _Float16 Wl[64 * 32];
  __shared__ __attribute__((aligned(16))) _Float16 Xl[32 * 128];  // frag-ready chunk
  __shared__ __attribute__((aligned(16))) float Yt[64 * 128];
  __shared__ float red[256];
  __shared__ float red2[256];
  const int tid = threadIdx.x, lane = tid & 31, wave = tid >> 5;
  const int colb = blockIdx.x * 128;
  const int rowb = blockIdx.y * 64;
  v8f acc[4];
#pragma unroll
  for (int mt = 0; mt < 4; ++mt)
#pragma unroll
    for (int j = 0; j < 8; ++j) acc[mt][j] = 0.f;

  for (int ks = 0; ks < C; ks += 32) {
    for (int i = tid; i < 64 * 32; i += 256) {
      int o = i >> 5, k = i & 31;
      Wl[i] = (_Float16)W[(size_t)(rowb + o) * C + ks + k];
    }
    for (int i = tid; i < 32 * 128; i += 256) {
      int k = i >> 7, c = i & 127;
      Xl[((((c << 1) + (k >> 4)) << 4)) + (k & 15)] = X[(size_t)(ks + k) * cols + colb + c];
    }
    __syncthreads();
    v16h bf = frag_b_fr(Xl, 0, wave * 16, lane);
#pragma unroll
    for (int mt = 0; mt < 4; ++mt) {
      v16h af = frag_a(Wl, 32, mt * 16, 0, lane);
      acc[mt] = wmma16(af, bf, acc[mt]);
    }
    __syncthreads();
  }

  const int hi = (lane >> 4) & 1;
  const int ln = lane & 15;
  const int clocal = wave * 16 + ln;
  const int col = colb + clocal;

  if (MODE == 0) {
#pragma unroll
    for (int mt = 0; mt < 4; ++mt)
#pragma unroll
      for (int r = 0; r < 8; ++r) {
        int row = mt * 16 + r + hi * 8;
        Yt[row * 128 + clocal] = acc[mt][r];
      }
    __syncthreads();
    {
      int row = tid >> 2, sub = tid & 3;
      float s = 0.f, q = 0.f;
      int c0 = sub * 32;
      for (int c = c0; c < c0 + 32; ++c) {
        float v = Yt[row * 128 + c];
        s += v; q += v * v;
      }
      red[tid] = s; red2[tid] = q;
    }
    __syncthreads();
    if (tid < 64) {
      float s = red[tid * 4] + red[tid * 4 + 1] + red[tid * 4 + 2] + red[tid * 4 + 3];
      float q = red2[tid * 4] + red2[tid * 4 + 1] + red2[tid * 4 + 2] + red2[tid * 4 + 3];
      partS[(size_t)(rowb + tid) * gridDim.x + blockIdx.x] = s;
      partQ[(size_t)(rowb + tid) * gridDim.x + blockIdx.x] = q;
    }
  } else {
#pragma unroll
    for (int mt = 0; mt < 4; ++mt)
#pragma unroll
      for (int r = 0; r < 8; ++r) {
        int row = mt * 16 + r + hi * 8;
        int ch = rowb + row;
        float v = acc[mt][r];
        if (MODE == 1 || MODE == 2) v = lrelu(v * sb[ch] + sb[O + ch]);
        if (MODE == 1)      outH[(size_t)ch * cols + col] = (_Float16)v;
        else if (MODE == 2) outF[(size_t)ch * cols + col] = v;
        else                outF[(size_t)col * O + ch] = v;   // transposed (B,N,O)
      }
  }
}

// =====================================================================================
extern "C" void kernel_launch(void* const* d_in, const int* in_sizes, int n_in,
                              void* d_out, int out_size, void* d_ws, size_t ws_size,
                              hipStream_t stream) {
  (void)in_sizes; (void)n_in; (void)out_size; (void)ws_size;
  const int Bb = 2, Nn = 4096, Mm = 512;
  const float* x     = (const float*)d_in[0];
  const float* xgrid = (const float*)d_in[1];
  const int*   FPS   = (const int*)d_in[2];
  const float* Wc[9];
  for (int j = 0; j < 9; ++j) Wc[j] = (const float*)d_in[3 + j];
  const float* g[8]; const float* bt[8];
  for (int j = 0; j < 8; ++j) { g[j] = (const float*)d_in[12 + 2 * j]; bt[j] = (const float*)d_in[13 + 2 * j]; }

  char* base = (char*)d_ws;
  size_t off = 0;
  auto alloc = [&](size_t bytes) -> void* {
    void* p = base + off;
    off += (bytes + 255) & ~(size_t)255;
    return p;
  };
  int* nbr1 = (int*)alloc((size_t)Bb * Mm * KNBR * 4);
  int* nbr2 = (int*)alloc((size_t)Bb * Mm * KNBR * 4);
  int* nbr3 = (int*)alloc((size_t)Bb * Mm * KNBR * 4);
  int* nr1  = (int*)alloc((size_t)Bb * Nn * 4);
  int* nr2  = (int*)alloc((size_t)Bb * Nn * 4);
  int* nr3  = (int*)alloc((size_t)Bb * Nn * 4);
  float* x1  = (float*)alloc((size_t)Bb * 64 * Nn * 4);
  float* x2  = (float*)alloc((size_t)Bb * 64 * Nn * 4);
  float* x3  = (float*)alloc((size_t)Bb * 64 * Nn * 4);
  float* x1g = (float*)alloc((size_t)Bb * 64 * Mm * 4);
  float* x2g = (float*)alloc((size_t)Bb * 64 * Mm * 4);
  float* x3g = (float*)alloc((size_t)Bb * 64 * Mm * 4);
  float* partS = (float*)alloc((size_t)262144 * 4);
  float* partQ = (float*)alloc((size_t)262144 * 4);
  float* sbv[8];
  for (int j = 0; j < 8; ++j) sbv[j] = (float*)alloc(2 * 512 * 4);
  _Float16* xg16  = (_Float16*)alloc((size_t)192 * Bb * Mm * 2);
  float*    y6    = (float*)alloc((size_t)512 * Bb * Mm * 4);
  float*    xgmax = (float*)alloc((size_t)Bb * 512 * 4);
  _Float16* hin16 = (_Float16*)alloc((size_t)704 * Bb * Nn * 2);
  _Float16* h1_16 = (_Float16*)alloc((size_t)128 * Bb * Nn * 2);
  _Float16* h2_16 = (_Float16*)alloc((size_t)64 * Bb * Nn * 2);

  const int TPB = 256;
  const int eblocks = Bb * Nn / 2;                 // 4096
  const float cntE = (float)(Bb * Nn * KNBR);      // 409600

  // ---- Stage 1 (CIN=3, KD=32): W1, W2 ----
  knn_kernel<<<(Bb * Mm + TPB - 1) / TPB, TPB, 0, stream>>>(xgrid, 3, Mm, Bb * Mm, nbr1);
  nearest_kernel<<<(Bb * Nn + TPB - 1) / TPB, TPB, 0, stream>>>(x, xgrid, 3, Nn, Mm, Bb * Nn, nr1);
  edgeconv_kernel<3, 32, false, false><<<eblocks, 256, 0, stream>>>(
      x, xgrid, nr1, nbr1, Wc[0], nullptr, nullptr, nullptr, partS, partQ, nullptr, Nn, Mm);
  reduce_stats_kernel<<<1, 64, 0, stream>>>(partS, partQ, eblocks, cntE, g[0], bt[0], sbv[0], 64);
  edgeconv_kernel<3, 32, true, false><<<eblocks, 256, 0, stream>>>(
      x, xgrid, nr1, nbr1, Wc[0], Wc[1], sbv[0], nullptr, partS, partQ, nullptr, Nn, Mm);
  reduce_stats_kernel<<<1, 64, 0, stream>>>(partS, partQ, eblocks, cntE, g[1], bt[1], sbv[1], 64);
  edgeconv_kernel<3, 32, true, true><<<eblocks, 256, 0, stream>>>(
      x, xgrid, nr1, nbr1, Wc[0], Wc[1], sbv[0], sbv[1], nullptr, nullptr, x1, Nn, Mm);
  fps_gather_kernel<<<(Bb * 64 * Mm + TPB - 1) / TPB, TPB, 0, stream>>>(x1, FPS, x1g, 64, Nn, Mm, Bb * 64 * Mm);

  // ---- Stage 2 (CIN=64, KD=128): W3, W4 ----
  knn_kernel<<<(Bb * Mm + TPB - 1) / TPB, TPB, 0, stream>>>(x1g, 64, Mm, Bb * Mm, nbr2);
  nearest_kernel<<<(Bb * Nn + TPB - 1) / TPB, TPB, 0, stream>>>(x1, x1g, 64, Nn, Mm, Bb * Nn, nr2);
  edgeconv_kernel<64, 128, false, false><<<eblocks, 256, 0, stream>>>(
      x1, x1g, nr2, nbr2, Wc[2], nullptr, nullptr, nullptr, partS, partQ, nullptr, Nn, Mm);
  reduce_stats_kernel<<<1, 64, 0, stream>>>(partS, partQ, eblocks, cntE, g[2], bt[2], sbv[2], 64);
  edgeconv_kernel<64, 128, true, false><<<eblocks, 256, 0, stream>>>(
      x1, x1g, nr2, nbr2, Wc[2], Wc[3], sbv[2], nullptr, partS, partQ, nullptr, Nn, Mm);
  reduce_stats_kernel<<<1, 64, 0, stream>>>(partS, partQ, eblocks, cntE, g[3], bt[3], sbv[3], 64);
  edgeconv_kernel<64, 128, true, true><<<eblocks, 256, 0, stream>>>(
      x1, x1g, nr2, nbr2, Wc[2], Wc[3], sbv[2], sbv[3], nullptr, nullptr, x2, Nn, Mm);
  fps_gather_kernel<<<(Bb * 64 * Mm + TPB - 1) / TPB, TPB, 0, stream>>>(x2, FPS, x2g, 64, Nn, Mm, Bb * 64 * Mm);

  // ---- Stage 3 (CIN=64, KD=128): W5 only ----
  knn_kernel<<<(Bb * Mm + TPB - 1) / TPB, TPB, 0, stream>>>(x2g, 64, Mm, Bb * Mm, nbr3);
  nearest_kernel<<<(Bb * Nn + TPB - 1) / TPB, TPB, 0, stream>>>(x2, x2g, 64, Nn, Mm, Bb * Nn, nr3);
  edgeconv_kernel<64, 128, false, false><<<eblocks, 256, 0, stream>>>(
      x2, x2g, nr3, nbr3, Wc[4], nullptr, nullptr, nullptr, partS, partQ, nullptr, Nn, Mm);
  reduce_stats_kernel<<<1, 64, 0, stream>>>(partS, partQ, eblocks, cntE, g[4], bt[4], sbv[4], 64);
  edgeconv_kernel<64, 128, false, true><<<eblocks, 256, 0, stream>>>(
      x2, x2g, nr3, nbr3, Wc[4], nullptr, sbv[4], nullptr, nullptr, nullptr, x3, Nn, Mm);
  fps_gather_kernel<<<(Bb * 64 * Mm + TPB - 1) / TPB, TPB, 0, stream>>>(x3, FPS, x3g, 64, Nn, Mm, Bb * 64 * Mm);

  // ---- Head: W6 over M cols ----
  const int colsM = Bb * Mm;     // 1024
  const int colsN = Bb * Nn;     // 8192
  build_xg_kernel<<<(192 * colsM + TPB - 1) / TPB, TPB, 0, stream>>>(x1g, x2g, x3g, xg16, Mm, Bb);
  conv1_kernel<0><<<dim3(colsM / 128, 512 / 64), 256, 0, stream>>>(
      Wc[5], xg16, 192, colsM, 512, nullptr, partS, partQ, nullptr, nullptr);
  reduce_stats_kernel<<<2, 256, 0, stream>>>(partS, partQ, colsM / 128, (float)colsM, g[5], bt[5], sbv[5], 512);
  conv1_kernel<2><<<dim3(colsM / 128, 512 / 64), 256, 0, stream>>>(
      Wc[5], xg16, 192, colsM, 512, sbv[5], nullptr, nullptr, nullptr, y6);
  max_m_kernel<<<(Bb * 512 + TPB - 1) / TPB, TPB, 0, stream>>>(y6, xgmax, Mm, Bb, 512);

  // ---- W7 (128 x 704), W8 (64 x 128), W9 (128 x 64, no BN) over N cols ----
  build_hin_kernel<<<(704 * colsN + TPB - 1) / TPB, TPB, 0, stream>>>(xgmax, x1, x2, x3, hin16, Nn, Bb);
  conv1_kernel<0><<<dim3(colsN / 128, 2), 256, 0, stream>>>(
      Wc[6], hin16, 704, colsN, 128, nullptr, partS, partQ, nullptr, nullptr);
  reduce_stats_kernel<<<1, 128, 0, stream>>>(partS, partQ, colsN / 128, (float)colsN, g[6], bt[6], sbv[6], 128);
  conv1_kernel<1><<<dim3(colsN / 128, 2), 256, 0, stream>>>(
      Wc[6], hin16, 704, colsN, 128, sbv[6], nullptr, nullptr, h1_16, nullptr);
  conv1_kernel<0><<<dim3(colsN / 128, 1), 256, 0, stream>>>(
      Wc[7], h1_16, 128, colsN, 64, nullptr, partS, partQ, nullptr, nullptr);
  reduce_stats_kernel<<<1, 64, 0, stream>>>(partS, partQ, colsN / 128, (float)colsN, g[7], bt[7], sbv[7], 64);
  conv1_kernel<1><<<dim3(colsN / 128, 1), 256, 0, stream>>>(
      Wc[7], h1_16, 128, colsN, 64, sbv[7], nullptr, nullptr, h2_16, nullptr);
  conv1_kernel<3><<<dim3(colsN / 128, 2), 256, 0, stream>>>(
      Wc[8], h2_16, 64, colsN, 128, nullptr, nullptr, nullptr, nullptr, (float*)d_out);
}
